// LSTM_6708738916706
// MI455X (gfx1250) — matrix operand order repaired and checked by reading, not compiled, not run on hardware
//
#include <hip/hip_runtime.h>
#include <hip/hip_bf16.h>
#include <cstdint>

// Problem sizes
static constexpr int kB  = 128;    // batch
static constexpr int kT  = 512;    // timesteps
static constexpr int kI  = 256;    // input features (layer 0)
static constexpr int kH  = 1024;   // hidden
static constexpr int kO  = 128;    // output features
static constexpr int k4H = 4096;   // 4*H gate rows
static constexpr int K0  = kI + kH;   // 1280 concat K, layer 0
static constexpr int K1  = kH + kH;   // 2048 concat K, layer 1

// Launch shape: persistent kernel
static constexpr int NWG = 64;     // one WG per WGP (256KB LDS each)
static constexpr int NTH = 256;    // 8 wave32 per WG
static constexpr int NH  = 16;     // hidden units owned per WG (64*16 = 1024)

typedef __bf16 bf16;
typedef __attribute__((ext_vector_type(16))) __bf16 v16bf;
typedef __attribute__((ext_vector_type(8)))  float  v8f;

static constexpr int LDS_BYTES = 64 * K1 * 2;  // 256 KB weight slab (max of both layers)

// ---------------------------------------------------------------------------
// Grid-wide barrier: monotonic counter, release add + acquire spin.
// Counter is hipMemsetAsync'ed to 0 at the start of every kernel_launch.
// ---------------------------------------------------------------------------
__device__ __forceinline__ void grid_barrier(unsigned* cnt, unsigned* barId) {
  __threadfence();
  __syncthreads();
  if (threadIdx.x == 0) {
    unsigned target = (++(*barId)) * (unsigned)NWG;
    __hip_atomic_fetch_add(cnt, 1u, __ATOMIC_ACQ_REL, __HIP_MEMORY_SCOPE_AGENT);
    while (__hip_atomic_load(cnt, __ATOMIC_ACQUIRE, __HIP_MEMORY_SCOPE_AGENT) < target)
      __builtin_amdgcn_s_sleep(8);
  }
  __syncthreads();
}

// ---------------------------------------------------------------------------
// Fill this WG's LDS weight slab (64 rows x K bf16) with async global->LDS
// copies (ASYNCcnt path). LDS row r corresponds to gate g=r/16, unit n0+(r%16),
// i.e. global row (r/16)*H + n0 + (r%16) of the concatenated [4H x K] matrix.
// ---------------------------------------------------------------------------
__device__ __forceinline__ void load_weights_lds(bf16* ldsw, const bf16* __restrict__ wcat,
                                                 int K, int n0, int tid) {
  uint32_t lbase = (uint32_t)(uintptr_t)ldsw;   // low 32 bits = LDS byte offset
  const int cpr = K >> 3;                       // 16B chunks per row
  const int total = 64 * cpr;
  for (int c = tid; c < total; c += NTH) {
    int r  = c / cpr;
    int cc = c - r * cpr;
    int grow = ((r >> 4) * kH) + n0 + (r & 15);
    const bf16* src = wcat + (size_t)grow * K + (cc << 3);
    uint32_t ldst = lbase + (uint32_t)((r * K + (cc << 3)) * 2);
    asm volatile("global_load_async_to_lds_b128 %0, %1, off"
                 :: "v"(ldst), "v"(src) : "memory");
  }
  asm volatile("s_wait_asynccnt 0" ::: "memory");
  __syncthreads();
}

__device__ __forceinline__ float sigf(float x) { return 1.0f / (1.0f + __expf(-x)); }

// ---------------------------------------------------------------------------
// One LSTM layer, 512 sequential steps. Per wave: 16 batch rows x 16 hidden
// units; 4 gate accumulators (i,f,g,o) as v8f; cell state `ct` stays in VGPRs
// for the entire sequence. SEQ=true: h ring buffer is the full sequence
// (slot t read, slot t+1 written). SEQ=false: 2-slot double buffer.
// ---------------------------------------------------------------------------
template<int KTOT, int IN, bool SEQ>
__device__ void layer_pass(const bf16* ldsw,
                           const bf16* __restrict__ xseq, size_t xstride,
                           bf16* __restrict__ hbuf,
                           const float* __restrict__ bi, const float* __restrict__ bh,
                           int n0, unsigned* cnt, unsigned* barId) {
  const int tid   = threadIdx.x;
  const int lane  = tid & 31;
  const int wv    = tid >> 5;
  const int m0    = wv * 16;          // batch-row block of this wave
  const int nlo   = lane & 15;        // N coordinate (unit) of this lane
  const int khalf = lane >> 4;        // A/B half-lane K selector
  const size_t BH = (size_t)kB * kH;

  float bias[4];
#pragma unroll
  for (int g = 0; g < 4; ++g)
    bias[g] = bi[g * kH + n0 + nlo] + bh[g * kH + n0 + nlo];

  v8f ct = {0.f, 0.f, 0.f, 0.f, 0.f, 0.f, 0.f, 0.f};  // cell state tile

  for (int t = 0; t < kT; ++t) {
    const bf16* xb = xseq + (size_t)t * xstride;
    const bf16* hr = SEQ ? (hbuf + (size_t)t * BH)       : (hbuf + (size_t)(t & 1) * BH);
    bf16*       hw = SEQ ? (hbuf + (size_t)(t + 1) * BH) : (hbuf + (size_t)((t + 1) & 1) * BH);

    v8f acc[4];
#pragma unroll
    for (int g = 0; g < 4; ++g) {
      v8f a;
#pragma unroll
      for (int r = 0; r < 8; ++r) a[r] = bias[g];   // bias depends only on N == lane
      acc[g] = a;
    }

    for (int kk = 0; kk < KTOT / 32; ++kk) {
      int kbase = kk * 32;
      const bf16* ab;
      int ld;
      if (kbase < IN) { ab = xb; ld = IN; }
      else            { ab = hr; ld = kH; kbase -= IN; }

      // A fragment: 16x32 bf16. Lane holds row m = m0 + (lane&15);
      // lanes 0-15 -> K {0..7,16..23}, lanes 16-31 -> K {8..15,24..31}.
      union { v16bf v; uint4 q[2]; } Af;
      const bf16* ap = ab + (size_t)(m0 + nlo) * ld + kbase + khalf * 8;
      Af.q[0] = *(const uint4*)ap;
      Af.q[1] = *(const uint4*)(ap + 16);

#pragma unroll
      for (int g = 0; g < 4; ++g) {
        // B fragment: 32x16 bf16. Lane holds weight row (column N = lane&15),
        // lanes 0-15 -> K {0..7,16..23}, lanes 16-31 -> K {8..15,24..31}.
        union { v16bf v; uint4 q[2]; } Bf;
        const bf16* bp = ldsw + (size_t)(g * 16 + nlo) * KTOT + kk * 32 + khalf * 8;
        Bf.q[0] = *(const uint4*)bp;
        Bf.q[1] = *(const uint4*)(bp + 16);
        acc[g] = __builtin_amdgcn_wmma_f32_16x16x32_bf16(
            false, Af.v, false, Bf.v, (short)0, acc[g], false, false);
      }
    }

    // Gate nonlinearities + cell/hidden update (lane-local: same (m,n) layout
    // for all four accumulators and ct).
#pragma unroll
    for (int r = 0; r < 8; ++r) {
      float iv = sigf(acc[0][r]);
      float fv = sigf(acc[1][r]);
      float gv = tanhf(acc[2][r]);
      float ov = sigf(acc[3][r]);
      float cv = fv * ct[r] + iv * gv;
      ct[r] = cv;
      float hv = ov * tanhf(cv);
      int mrow = m0 + r + khalf * 8;                  // C/D layout: M = r + 8*(lane>=16)
      hw[(size_t)mrow * kH + n0 + nlo] = (bf16)hv;
    }

    grid_barrier(cnt, barId);   // make h_t visible to all WGPs before step t+1
  }
}

// ---------------------------------------------------------------------------
// Persistent kernel: phase0 layout/precision conversion -> layer0 -> layer1 -> FC
// ---------------------------------------------------------------------------
__global__ __launch_bounds__(NTH, 1)
void lstm_persistent(const float* __restrict__ x,
                     const float* __restrict__ Wih0, const float* __restrict__ Whh0,
                     const float* __restrict__ bih0, const float* __restrict__ bhh0,
                     const float* __restrict__ Wih1, const float* __restrict__ Whh1,
                     const float* __restrict__ bih1, const float* __restrict__ bhh1,
                     const float* __restrict__ Wfc,  const float* __restrict__ bfc,
                     float* __restrict__ out,
                     bf16* __restrict__ xt,     // [T][B][I] bf16 (time-major)
                     bf16* __restrict__ wcat0,  // [4H][K0] bf16 = [Wih0 | Whh0]
                     bf16* __restrict__ wcat1,  // [4H][K1] bf16 = [Wih1 | Whh1]
                     bf16* __restrict__ hseq,   // [(T+1)][B][H] bf16, slot0 = zeros
                     bf16* __restrict__ hdb,    // [2][B][H] bf16 double buffer
                     unsigned* cnt) {
  extern __shared__ char smem[];
  bf16* ldsw = (bf16*)smem;

  const int tid  = threadIdx.x;
  const int wg   = blockIdx.x;
  const long gid = (long)wg * NTH + tid;
  const long gst = (long)NWG * NTH;
  unsigned barId = 0;

  // ---- Phase 0: conversions / layout (fully parallel) ----
  for (long i = gid; i < (long)kT * kB * kI; i += gst) {     // x -> time-major bf16
    int ii = (int)(i % kI);
    long rem = i / kI;
    int b = (int)(rem % kB);
    int t = (int)(rem / kB);
    xt[i] = (bf16)x[((size_t)b * kT + t) * kI + ii];
  }
  for (long i = gid; i < (long)k4H * K0; i += gst) {         // [Wih0|Whh0] bf16
    int col = (int)(i % K0);
    long row = i / K0;
    float v = (col < kI) ? Wih0[row * kI + col] : Whh0[row * kH + (col - kI)];
    wcat0[i] = (bf16)v;
  }
  for (long i = gid; i < (long)k4H * K1; i += gst) {         // [Wih1|Whh1] bf16
    int col = (int)(i % K1);
    long row = i / K1;
    float v = (col < kH) ? Wih1[row * kH + col] : Whh1[row * kH + (col - kH)];
    wcat1[i] = (bf16)v;
  }
  for (long i = gid; i < (long)kB * kH; i += gst) {          // zero h/c initials
    hseq[i] = (bf16)0.f;
    hdb[i] = (bf16)0.f;
    hdb[i + (size_t)kB * kH] = (bf16)0.f;
  }
  grid_barrier(cnt, &barId);

  const int n0 = wg * NH;   // hidden units owned by this WG

  // ---- Layer 0: K = I + H = 1280, x from xt, h sequence kept for layer 1 ----
  load_weights_lds(ldsw, wcat0, K0, n0, tid);
  layer_pass<K0, kI, true>(ldsw, xt, (size_t)kB * kI, hseq, bih0, bhh0, n0, cnt, &barId);

  // ---- Layer 1: K = 2H = 2048, x = hseq slots 1..T, h double-buffered ----
  load_weights_lds(ldsw, wcat1, K1, n0, tid);
  layer_pass<K1, kH, false>(ldsw, hseq + (size_t)kB * kH, (size_t)kB * kH,
                            hdb, bih1, bhh1, n0, cnt, &barId);

  // ---- FC epilogue: out[b,o] = b_fc[o] + h2_last . W_fc[o,:] ----
  // T even -> final h2 lives in hdb slot 0. Each WG covers 2 output columns.
  {
    const bf16* h2 = hdb;                  // slot 0
    int b = tid & 127;
    int o = wg * 2 + (tid >> 7);
    float s = bfc[o];
    for (int n = 0; n < kH; ++n)
      s += (float)h2[(size_t)b * kH + n] * Wfc[(size_t)o * kH + n];
    out[(size_t)b * kO + o] = s;
  }
}

// ---------------------------------------------------------------------------
extern "C" void kernel_launch(void* const* d_in, const int* in_sizes, int n_in,
                              void* d_out, int out_size, void* d_ws, size_t ws_size,
                              hipStream_t stream) {
  const float* x    = (const float*)d_in[0];
  const float* Wih0 = (const float*)d_in[1];
  const float* Whh0 = (const float*)d_in[2];
  const float* bih0 = (const float*)d_in[3];
  const float* bhh0 = (const float*)d_in[4];
  const float* Wih1 = (const float*)d_in[5];
  const float* Whh1 = (const float*)d_in[6];
  const float* bih1 = (const float*)d_in[7];
  const float* bhh1 = (const float*)d_in[8];
  const float* Wfc  = (const float*)d_in[9];
  const float* bfc  = (const float*)d_in[10];
  float* out = (float*)d_out;

  char* ws = (char*)d_ws;
  size_t off = 0;
  auto carve = [&](size_t bytes) -> void* {
    void* p = ws + off;
    off += (bytes + 255) & ~(size_t)255;
    return p;
  };
  bf16* xt    = (bf16*)carve((size_t)kT * kB * kI * 2);        // 32 MB
  bf16* wcat0 = (bf16*)carve((size_t)k4H * K0 * 2);            // 10 MB
  bf16* wcat1 = (bf16*)carve((size_t)k4H * K1 * 2);            // 16 MB
  bf16* hseq  = (bf16*)carve((size_t)(kT + 1) * kB * kH * 2);  // 128 MB
  bf16* hdb   = (bf16*)carve((size_t)2 * kB * kH * 2);         // 0.5 MB
  unsigned* cnt = (unsigned*)carve(256);

  // Monotonic barrier counter must start at 0 every replay (capture-legal).
  hipMemsetAsync(cnt, 0, 256, stream);

  hipFuncSetAttribute((const void*)lstm_persistent,
                      hipFuncAttributeMaxDynamicSharedMemorySize, LDS_BYTES);

  lstm_persistent<<<dim3(NWG), dim3(NTH), LDS_BYTES, stream>>>(
      x, Wih0, Whh0, bih0, bhh0, Wih1, Whh1, bih1, bhh1, Wfc, bfc,
      out, xt, wcat0, wcat1, hseq, hdb, cnt);
}
